// PointNet2Segmentation_57827439674260
// MI455X (gfx1250) — matrix operand order repaired and checked by reading, not compiled, not run on hardware
//
#include <hip/hip_runtime.h>

typedef __attribute__((ext_vector_type(16))) _Float16 v16h;
typedef __attribute__((ext_vector_type(8)))  float    v8f;

#define INFF __builtin_inff()

__device__ inline v8f wmma_f16(v16h a, v16h b, v8f c) {
  return __builtin_amdgcn_wmma_f32_16x16x32_f16(false, a, false, b, (short)0, c, false, false);
}

// A operand: 16 rows (m0..m0+15) x 32 K, from row-major f16, leading dim lda (halves).
// lane L: row = m0 + (L&15); halves 0..7 = K k0+kb+0..7, halves 8..15 = K k0+16+kb+0..7, kb=(L>=16)?8:0
__device__ inline v16h load_matA(const _Float16* base, int m0, int lda, int k0) {
  int lane = threadIdx.x & 31;
  int row  = m0 + (lane & 15);
  int kb   = k0 + ((lane >> 4) << 3);
  const _Float16* p = base + (size_t)row * lda + kb;
  v16h v;
  *((float4*)&v)     = *(const float4*)p;
  *((float4*)&v + 1) = *(const float4*)(p + 16);
  return v;
}

// B operand: 16 cols (n0..n0+15) x 32 K, taken from W rows (out = act @ W^T).
// lane L: col = n0 + (L&15); halves 0..7 = K k0+kb+0..7, halves 8..15 = K k0+16+kb+0..7, kb=(L>=16)?8:0
__device__ inline v16h load_matB(const _Float16* base, int n0, int ldb, int k0) {
  int lane = threadIdx.x & 31;
  int row  = n0 + (lane & 15);
  int kb   = k0 + ((lane >> 4) << 3);
  const _Float16* p = base + (size_t)row * ldb + kb;
  v16h v;
  *((float4*)&v)     = *(const float4*)p;
  *((float4*)&v + 1) = *(const float4*)(p + 16);
  return v;
}

// ---------------------------------------------------------------- prep kernels
__global__ void convpad_kernel(const float* __restrict__ W, int N, int K,
                               int Npad, int Kpad, _Float16* __restrict__ out) {
  int e = blockIdx.x * blockDim.x + threadIdx.x;
  if (e >= Npad * Kpad) return;
  int n = e / Kpad, k = e % Kpad;
  float v = (n < N && k < K) ? W[(size_t)n * K + k] : 0.f;
  out[e] = (_Float16)v;
}

__global__ void bnprep_kernel(const float* __restrict__ g, const float* __restrict__ b,
                              const float* __restrict__ m, const float* __restrict__ v,
                              float* __restrict__ alpha, float* __restrict__ beta) {
  int i = blockIdx.x * blockDim.x + threadIdx.x;
  if (i >= 512) return;
  float a = g[i] * rsqrtf(v[i] + 1e-5f);
  alpha[i] = a;
  beta[i]  = b[i] - m[i] * a;
}

// ---------------------------------------------------------------- FPS (1 block, 1024 threads)
__global__ void fps_kernel(const float* __restrict__ pos, int Np, int nSamples,
                           int* __restrict__ idx_out, float* __restrict__ spos) {
  __shared__ float rv[32];
  __shared__ int   ri[32];
  __shared__ float bpt[3];
  __shared__ int   sidx[4096];
  int tid = threadIdx.x, lane = tid & 31, wid = tid >> 5;
  int nth = blockDim.x;
  int npt = Np / nth;                 // 16 or 4
  float d[16];
  for (int j = 0; j < npt; ++j) d[j] = INFF;
  int last = 0;
  for (int t = 0; t < nSamples; ++t) {
    if (tid == 0) {
      sidx[t] = last;
      bpt[0] = pos[3 * last]; bpt[1] = pos[3 * last + 1]; bpt[2] = pos[3 * last + 2];
    }
    __syncthreads();
    float lx = bpt[0], ly = bpt[1], lz = bpt[2];
    float bestv = -1.f; int besti = 0;
    for (int j = 0; j < npt; ++j) {
      int i = tid + j * nth;
      float dx = pos[3 * i] - lx, dy = pos[3 * i + 1] - ly, dz = pos[3 * i + 2] - lz;
      float dd = dx * dx + dy * dy + dz * dz;
      d[j] = fminf(d[j], dd);
      if (d[j] > bestv) { bestv = d[j]; besti = i; }
    }
    for (int off = 16; off > 0; off >>= 1) {     // wave argmax (lowest index on tie)
      float ov = __shfl_down(bestv, off, 32);
      int   oi = __shfl_down(besti, off, 32);
      if (ov > bestv || (ov == bestv && oi < besti)) { bestv = ov; besti = oi; }
    }
    if (lane == 0) { rv[wid] = bestv; ri[wid] = besti; }
    __syncthreads();
    if (wid == 0) {
      float v2 = rv[lane]; int i2 = ri[lane];
      for (int off = 16; off > 0; off >>= 1) {
        float ov = __shfl_down(v2, off, 32);
        int   oi = __shfl_down(i2, off, 32);
        if (ov > v2 || (ov == v2 && oi < i2)) { v2 = ov; i2 = oi; }
      }
      if (lane == 0) ri[0] = i2;
    }
    __syncthreads();
    last = ri[0];
    __syncthreads();
  }
  for (int s = tid; s < nSamples; s += nth) {
    int i = sidx[s];
    idx_out[s] = i;
    spos[3 * s] = pos[3 * i]; spos[3 * s + 1] = pos[3 * i + 1]; spos[3 * s + 2] = pos[3 * i + 2];
  }
}

// ---------------------------------------------------------------- radius-kNN top-64 (1 WG / center)
__global__ void knn_kernel(const float* __restrict__ centers,
                           const float* __restrict__ src, int Ns,
                           float r2, int K, int* __restrict__ nbr) {
  extern __shared__ float d2arr[];           // Ns floats
  __shared__ float lmin[128]; __shared__ int lidx[128];
  __shared__ float rmv[128];  __shared__ int rmi[128];
  int c = blockIdx.x, tid = threadIdx.x;
  int slice = Ns / 128;
  float cx = centers[3 * c], cy = centers[3 * c + 1], cz = centers[3 * c + 2];
  int i0 = tid * slice;
  float lv = INFF; int li = i0;
  for (int j = 0; j < slice; ++j) {
    int i = i0 + j;
    float dx = src[3 * i] - cx, dy = src[3 * i + 1] - cy, dz = src[3 * i + 2] - cz;
    float dd = dx * dx + dy * dy + dz * dz;
    float v = (dd <= r2) ? dd : INFF;
    d2arr[i] = v;
    if (v < lv) { lv = v; li = i; }
  }
  lmin[tid] = lv; lidx[tid] = li;
  __syncthreads();
  for (int k = 0; k < K; ++k) {
    rmv[tid] = lmin[tid]; rmi[tid] = lidx[tid];
    __syncthreads();
    for (int s = 64; s > 0; s >>= 1) {
      if (tid < s) {
        float ov = rmv[tid + s]; int oi = rmi[tid + s];
        if (ov < rmv[tid] || (ov == rmv[tid] && oi < rmi[tid])) { rmv[tid] = ov; rmi[tid] = oi; }
      }
      __syncthreads();
    }
    float bv = rmv[0]; int bi = rmi[0];
    if (tid == 0) nbr[(size_t)c * K + k] = (bv < INFF) ? bi : -1;
    if (bv < INFF && tid == (bi / slice)) {    // owner excludes + rescans its slice
      d2arr[bi] = INFF;
      float nv = INFF; int ni = i0;
      for (int j = 0; j < slice; ++j) { float v = d2arr[i0 + j]; if (v < nv) { nv = v; ni = i0 + j; } }
      lmin[tid] = nv; lidx[tid] = ni;
    }
    __syncthreads();
  }
}

// ---------------------------------------------------------------- point-conv (1 WG / center, 4 waves)
template<int K0PAD, int XDIM, int H1, int H2, int H3>
__global__ void pointconv_kernel(const float* __restrict__ centers,
                                 const float* __restrict__ srcpos,
                                 const float* __restrict__ srcfeat,
                                 const int* __restrict__ nbr,
                                 const _Float16* __restrict__ W0,
                                 const _Float16* __restrict__ W1,
                                 const _Float16* __restrict__ W2,
                                 const float* __restrict__ b0,
                                 const float* __restrict__ b1,
                                 const float* __restrict__ b2,
                                 float* __restrict__ xout) {
  __shared__ _Float16 feat0[64][K0PAD];
  __shared__ _Float16 act1[64][H1];
  __shared__ _Float16 act2[64][H2];
  __shared__ float validf[64];
  __shared__ float wavemax[4][H3];
  int c = blockIdx.x, tid = threadIdx.x;
  float cx = centers[3 * c], cy = centers[3 * c + 1], cz = centers[3 * c + 2];
  if (tid < 64) {
    int id = nbr[(size_t)c * 64 + tid];
    validf[tid] = (id >= 0) ? 1.f : 0.f;
    int s = (id >= 0) ? id : 0;
    int k = 0;
    if (XDIM > 0)
      for (; k < XDIM; ++k) feat0[tid][k] = (_Float16)srcfeat[(size_t)s * XDIM + k];
    feat0[tid][k + 0] = (_Float16)(srcpos[3 * s]     - cx);
    feat0[tid][k + 1] = (_Float16)(srcpos[3 * s + 1] - cy);
    feat0[tid][k + 2] = (_Float16)(srcpos[3 * s + 2] - cz);
    for (int kk = k + 3; kk < K0PAD; ++kk) feat0[tid][kk] = (_Float16)0.f;
  }
  __syncthreads();
  int wave = tid >> 5, lane = tid & 31, lh = lane >> 4;
  int m0 = wave * 16;
  // L1: feat0 (K0PAD) -> act1 (H1), relu
  for (int nt = 0; nt < H1 / 16; ++nt) {
    v8f acc = {};
    for (int k0 = 0; k0 < K0PAD; k0 += 32)
      acc = wmma_f16(load_matA(&feat0[0][0], m0, K0PAD, k0), load_matB(W0, nt * 16, K0PAD, k0), acc);
    int n = nt * 16 + (lane & 15);
    float bn = b0[n];
#pragma unroll
    for (int r = 0; r < 8; ++r) act1[m0 + r + 8 * lh][n] = (_Float16)fmaxf(acc[r] + bn, 0.f);
  }
  __syncthreads();
  // L2: act1 (H1) -> act2 (H2), relu
  for (int nt = 0; nt < H2 / 16; ++nt) {
    v8f acc = {};
    for (int k0 = 0; k0 < H1; k0 += 32)
      acc = wmma_f16(load_matA(&act1[0][0], m0, H1, k0), load_matB(W1, nt * 16, H1, k0), acc);
    int n = nt * 16 + (lane & 15);
    float bn = b1[n];
#pragma unroll
    for (int r = 0; r < 8; ++r) act2[m0 + r + 8 * lh][n] = (_Float16)fmaxf(acc[r] + bn, 0.f);
  }
  __syncthreads();
  // L3: act2 (H2) -> H3, bias only, masked max over the 64 neighbor rows
  for (int nt = 0; nt < H3 / 16; ++nt) {
    v8f acc = {};
    for (int k0 = 0; k0 < H2; k0 += 32)
      acc = wmma_f16(load_matA(&act2[0][0], m0, H2, k0), load_matB(W2, nt * 16, H2, k0), acc);
    int n = nt * 16 + (lane & 15);
    float bn = b2[n];
    float mx = -INFF;
#pragma unroll
    for (int r = 0; r < 8; ++r) {
      int m = m0 + r + 8 * lh;
      float v = acc[r] + bn;
      mx = fmaxf(mx, (validf[m] > 0.f) ? v : -INFF);
    }
    mx = fmaxf(mx, __shfl_xor(mx, 16, 32));
    if (lane < 16) wavemax[wave][nt * 16 + lane] = mx;
  }
  __syncthreads();
  for (int t = tid; t < H3; t += blockDim.x) {
    float m = fmaxf(fmaxf(wavemax[0][t], wavemax[1][t]), fmaxf(wavemax[2][t], wavemax[3][t]));
    xout[(size_t)c * H3 + t] = m;
  }
}

// ---------------------------------------------------------------- generic WMMA GEMM: out = epi(A @ B^T)
__global__ void wmma_gemm_kernel(const _Float16* __restrict__ A, int lda,
                                 const _Float16* __restrict__ B, int ldb,
                                 const float* __restrict__ bias,
                                 const float* __restrict__ alpha,
                                 const float* __restrict__ beta,
                                 int M, int Nreal, int Kpad,
                                 _Float16* __restrict__ outH, int ldoh,
                                 float* __restrict__ outF, int ldof,
                                 int relu, int transposeF) {
  int wave = threadIdx.x >> 5, lane = threadIdx.x & 31, lh = lane >> 4;
  int m0 = blockIdx.x * 64 + wave * 16;
  int Npad = (Nreal + 15) & ~15;
  for (int nt = 0; nt < 4; ++nt) {
    int n0 = blockIdx.y * 64 + nt * 16;
    if (n0 >= Npad) break;
    v8f acc = {};
    for (int k0 = 0; k0 < Kpad; k0 += 32)
      acc = wmma_f16(load_matA(A, m0, lda, k0), load_matB(B, n0, ldb, k0), acc);
    int n = n0 + (lane & 15);
    if (n < Nreal) {
      float bn = bias  ? bias[n]  : 0.f;
      float al = alpha ? alpha[n] : 1.f;
      float be = beta  ? beta[n]  : 0.f;
#pragma unroll
      for (int r = 0; r < 8; ++r) {
        int m = m0 + r + 8 * lh;
        float v = (acc[r] + bn) * al + be;
        if (relu) v = fmaxf(v, 0.f);
        if (outH) outH[(size_t)m * ldoh + n] = (_Float16)v;
        if (outF) {
          if (transposeF) outF[(size_t)n * ldof + m] = v;
          else            outF[(size_t)m * ldof + n] = v;
        }
      }
    }
  }
}

// ---------------------------------------------------------------- misc small kernels
__global__ void colmax_kernel(const float* __restrict__ X, int M, int N, float* __restrict__ g) {
  int c = blockIdx.x * blockDim.x + threadIdx.x;
  if (c >= N) return;
  float m = -INFF;
  for (int r = 0; r < M; ++r) m = fmaxf(m, X[(size_t)r * N + c]);
  g[c] = m;
}

__global__ void build_a3_kernel(const float* __restrict__ x2, const float* __restrict__ pos2,
                                _Float16* __restrict__ A) {
  int e = blockIdx.x * blockDim.x + threadIdx.x;
  if (e >= 1024 * 288) return;
  int m = e / 288, k = e % 288;
  float v = (k < 256) ? x2[(size_t)m * 256 + k] : (k < 259 ? pos2[3 * m + (k - 256)] : 0.f);
  A[e] = (_Float16)v;
}

__global__ void build_u1A_kernel(const float* __restrict__ x2, const float* __restrict__ g,
                                 _Float16* __restrict__ A) {
  int e = blockIdx.x * blockDim.x + threadIdx.x;
  if (e >= 1024 * 1280) return;
  int m = e / 1280, k = e % 1280;
  float v = (k < 256) ? x2[(size_t)m * 256 + k] : g[k - 256];   // f3to2 == g broadcast (single source pt)
  A[e] = (_Float16)v;
}

__global__ void interp_idx_kernel(const float* __restrict__ to_pos, int Nt,
                                  const float* __restrict__ from_pos, int Nf,
                                  int* __restrict__ idx3, float* __restrict__ w3) {
  int t = blockIdx.x * blockDim.x + threadIdx.x;
  if (t >= Nt) return;
  float px = to_pos[3 * t], py = to_pos[3 * t + 1], pz = to_pos[3 * t + 2];
  float d0 = INFF, d1 = INFF, d2v = INFF; int i0 = 0, i1 = 0, i2 = 0;
  for (int i = 0; i < Nf; ++i) {
    float dx = from_pos[3 * i] - px, dy = from_pos[3 * i + 1] - py, dz = from_pos[3 * i + 2] - pz;
    float dd = dx * dx + dy * dy + dz * dz;
    if (dd < d0)      { d2v = d1; i2 = i1; d1 = d0; i1 = i0; d0 = dd; i0 = i; }
    else if (dd < d1) { d2v = d1; i2 = i1; d1 = dd; i1 = i; }
    else if (dd < d2v){ d2v = dd; i2 = i; }
  }
  float w0 = 1.f / (d0 + 1e-8f), w1 = 1.f / (d1 + 1e-8f), w2 = 1.f / (d2v + 1e-8f);
  float s = w0 + w1 + w2;
  idx3[3 * t] = i0; idx3[3 * t + 1] = i1; idx3[3 * t + 2] = i2;
  w3[3 * t] = w0 / s; w3[3 * t + 1] = w1 / s; w3[3 * t + 2] = w2 / s;
}

// A row m: [ pre[m] (preC cols) | sum_j w_j * src[idx_j] (C cols) ]
__global__ void build_interpA_kernel(const float* __restrict__ pre, int preC,
                                     const _Float16* __restrict__ src, int C,
                                     const int* __restrict__ idx3, const float* __restrict__ w3,
                                     int Nt, _Float16* __restrict__ out) {
  int ld = preC + C;
  int e = blockIdx.x * blockDim.x + threadIdx.x;
  if (e >= Nt * ld) return;
  int m = e / ld, k = e % ld;
  float v;
  if (k < preC) v = pre[(size_t)m * preC + k];
  else {
    int kk = k - preC;
    int j0 = idx3[3 * m], j1 = idx3[3 * m + 1], j2 = idx3[3 * m + 2];
    v = w3[3 * m]     * (float)src[(size_t)j0 * C + kk]
      + w3[3 * m + 1] * (float)src[(size_t)j1 * C + kk]
      + w3[3 * m + 2] * (float)src[(size_t)j2 * C + kk];
  }
  out[e] = (_Float16)v;
}

// ---------------------------------------------------------------- host side
static inline void launch_gemm(const _Float16* A, int lda, const _Float16* B, int ldb,
                               const float* bias, const float* alpha, const float* beta,
                               int M, int Nreal, int Kpad,
                               _Float16* outH, int ldoh, float* outF, int ldof,
                               int relu, int trans, hipStream_t s) {
  int Npad = (Nreal + 15) & ~15;
  dim3 grid(M / 64, (Npad + 63) / 64);
  wmma_gemm_kernel<<<grid, 128, 0, s>>>(A, lda, B, ldb, bias, alpha, beta, M, Nreal, Kpad,
                                        outH, ldoh, outF, ldof, relu, trans);
}

static inline void launch_convpad(const float* W, int N, int K, int Npad, int Kpad,
                                  _Float16* out, hipStream_t s) {
  int n = Npad * Kpad;
  convpad_kernel<<<(n + 255) / 256, 256, 0, s>>>(W, N, K, Npad, Kpad, out);
}

extern "C" void kernel_launch(void* const* d_in, const int* in_sizes, int n_in,
                              void* d_out, int out_size, void* d_ws, size_t ws_size,
                              hipStream_t stream) {
  (void)in_sizes; (void)n_in; (void)out_size; (void)ws_size;
  const float* pos   = (const float*)d_in[0];
  const float* m1w0  = (const float*)d_in[1];
  const float* m1w1  = (const float*)d_in[2];
  const float* m1w2  = (const float*)d_in[3];
  const float* m1b0  = (const float*)d_in[4];
  const float* m1b1  = (const float*)d_in[5];
  const float* m1b2  = (const float*)d_in[6];
  const float* m2w0  = (const float*)d_in[7];
  const float* m2w1  = (const float*)d_in[8];
  const float* m2w2  = (const float*)d_in[9];
  const float* m2b0  = (const float*)d_in[10];
  const float* m2b1  = (const float*)d_in[11];
  const float* m2b2  = (const float*)d_in[12];
  const float* m3w0  = (const float*)d_in[13];
  const float* m3w1  = (const float*)d_in[14];
  const float* m3w2  = (const float*)d_in[15];
  const float* m3b0  = (const float*)d_in[16];
  const float* m3b1  = (const float*)d_in[17];
  const float* m3b2  = (const float*)d_in[18];
  const float* up1w  = (const float*)d_in[19];
  const float* up1b  = (const float*)d_in[20];
  const float* up2w  = (const float*)d_in[21];
  const float* up2b  = (const float*)d_in[22];
  const float* seg1w = (const float*)d_in[23];
  const float* seg1b = (const float*)d_in[24];
  const float* bng   = (const float*)d_in[25];
  const float* bnb   = (const float*)d_in[26];
  const float* bnm   = (const float*)d_in[27];
  const float* bnv   = (const float*)d_in[28];
  const float* seg2w = (const float*)d_in[29];
  const float* seg2b = (const float*)d_in[30];

  char* p = (char*)d_ws;
  auto alloc = [&](size_t bytes) -> void* {
    void* r = (void*)p;
    p += (bytes + 255) & ~(size_t)255;
    return r;
  };
  // f16 padded weights
  _Float16* w10h = (_Float16*)alloc(64  * 32   * 2);
  _Float16* w11h = (_Float16*)alloc(64  * 64   * 2);
  _Float16* w12h = (_Float16*)alloc(128 * 64   * 2);
  _Float16* w20h = (_Float16*)alloc(128 * 160  * 2);
  _Float16* w21h = (_Float16*)alloc(128 * 128  * 2);
  _Float16* w22h = (_Float16*)alloc(256 * 128  * 2);
  _Float16* w30h = (_Float16*)alloc(256 * 288  * 2);
  _Float16* w31h = (_Float16*)alloc(512 * 256  * 2);
  _Float16* w32h = (_Float16*)alloc((size_t)1024 * 512 * 2);
  _Float16* up1h = (_Float16*)alloc((size_t)512 * 1280 * 2);
  _Float16* up2h = (_Float16*)alloc((size_t)256 * 640  * 2);
  _Float16* sg1h = (_Float16*)alloc((size_t)512 * 256  * 2);
  _Float16* sg2h = (_Float16*)alloc((size_t)16  * 512  * 2);
  float* alphaB  = (float*)alloc(512 * 4);
  float* betaB   = (float*)alloc(512 * 4);
  // pipeline buffers
  int*   idx1 = (int*)  alloc(4096 * 4);
  float* pos1 = (float*)alloc(4096 * 3 * 4);
  int*   nbr1 = (int*)  alloc((size_t)4096 * 64 * 4);
  float* x1   = (float*)alloc((size_t)4096 * 128 * 4);
  int*   idx2 = (int*)  alloc(1024 * 4);
  float* pos2 = (float*)alloc(1024 * 3 * 4);
  int*   nbr2 = (int*)  alloc((size_t)1024 * 64 * 4);
  float* x2   = (float*)alloc((size_t)1024 * 256 * 4);
  _Float16* A3h  = (_Float16*)alloc((size_t)1024 * 288 * 2);
  _Float16* a31h = (_Float16*)alloc((size_t)1024 * 256 * 2);
  _Float16* a32h = (_Float16*)alloc((size_t)1024 * 512 * 2);
  float*    a33f = (float*)   alloc((size_t)1024 * 1024 * 4);
  float*    gmax = (float*)   alloc(1024 * 4);
  _Float16* Au1h = (_Float16*)alloc((size_t)1024 * 1280 * 2);
  _Float16* u1h  = (_Float16*)alloc((size_t)1024 * 512 * 2);
  int*      iw1i = (int*)     alloc((size_t)4096 * 3 * 4);
  float*    iw1w = (float*)   alloc((size_t)4096 * 3 * 4);
  _Float16* Au2h = (_Float16*)alloc((size_t)4096 * 640 * 2);
  _Float16* u2h  = (_Float16*)alloc((size_t)4096 * 256 * 2);
  int*      iw0i = (int*)     alloc((size_t)16384 * 3 * 4);
  float*    iw0w = (float*)   alloc((size_t)16384 * 3 * 4);
  _Float16* Afh  = (_Float16*)alloc((size_t)16384 * 256 * 2);
  _Float16* hh   = (_Float16*)alloc((size_t)16384 * 512 * 2);

  // -------- prep: weights -> f16 padded, BN folding
  launch_convpad(m1w0,  64,   3,   64, 32,   w10h, stream);
  launch_convpad(m1w1,  64,   64,  64, 64,   w11h, stream);
  launch_convpad(m1w2,  128,  64, 128, 64,   w12h, stream);
  launch_convpad(m2w0,  128, 131, 128, 160,  w20h, stream);
  launch_convpad(m2w1,  128, 128, 128, 128,  w21h, stream);
  launch_convpad(m2w2,  256, 128, 256, 128,  w22h, stream);
  launch_convpad(m3w0,  256, 259, 256, 288,  w30h, stream);
  launch_convpad(m3w1,  512, 256, 512, 256,  w31h, stream);
  launch_convpad(m3w2, 1024, 512, 1024, 512, w32h, stream);
  launch_convpad(up1w,  512, 1280, 512, 1280, up1h, stream);
  launch_convpad(up2w,  256, 640, 256, 640,  up2h, stream);
  launch_convpad(seg1w, 512, 256, 512, 256,  sg1h, stream);
  launch_convpad(seg2w, 13,  512, 16,  512,  sg2h, stream);
  bnprep_kernel<<<2, 256, 0, stream>>>(bng, bnb, bnm, bnv, alphaB, betaB);

  // -------- SA level 1
  fps_kernel<<<1, 1024, 0, stream>>>(pos, 16384, 4096, idx1, pos1);
  knn_kernel<<<4096, 128, 16384 * 4, stream>>>(pos1, pos, 16384, 0.2f * 0.2f, 64, nbr1);
  pointconv_kernel<32, 0, 64, 64, 128><<<4096, 128, 0, stream>>>(
      pos1, pos, nullptr, nbr1, w10h, w11h, w12h, m1b0, m1b1, m1b2, x1);

  // -------- SA level 2
  fps_kernel<<<1, 1024, 0, stream>>>(pos1, 4096, 1024, idx2, pos2);
  knn_kernel<<<1024, 128, 4096 * 4, stream>>>(pos2, pos1, 4096, 0.4f * 0.4f, 64, nbr2);
  pointconv_kernel<160, 128, 128, 128, 256><<<1024, 128, 0, stream>>>(
      pos2, pos1, x1, nbr2, w20h, w21h, w22h, m2b0, m2b1, m2b2, x2);

  // -------- global MLP3 + max
  build_a3_kernel<<<(1024 * 288 + 255) / 256, 256, 0, stream>>>(x2, pos2, A3h);
  launch_gemm(A3h, 288, w30h, 288, m3b0, nullptr, nullptr, 1024, 256, 288, a31h, 256, nullptr, 0, 1, 0, stream);
  launch_gemm(a31h, 256, w31h, 256, m3b1, nullptr, nullptr, 1024, 512, 256, a32h, 512, nullptr, 0, 1, 0, stream);
  launch_gemm(a32h, 512, w32h, 512, m3b2, nullptr, nullptr, 1024, 1024, 512, nullptr, 0, a33f, 1024, 0, 0, stream);
  colmax_kernel<<<4, 256, 0, stream>>>(a33f, 1024, 1024, gmax);

  // -------- FP: up1 (f3to2 == g broadcast)
  build_u1A_kernel<<<(1024 * 1280 + 255) / 256, 256, 0, stream>>>(x2, gmax, Au1h);
  launch_gemm(Au1h, 1280, up1h, 1280, up1b, nullptr, nullptr, 1024, 512, 1280, u1h, 512, nullptr, 0, 0, 0, stream);

  // -------- FP: interp pos2->pos1, up2
  interp_idx_kernel<<<(4096 + 255) / 256, 256, 0, stream>>>(pos1, 4096, pos2, 1024, iw1i, iw1w);
  build_interpA_kernel<<<(4096 * 640 + 255) / 256, 256, 0, stream>>>(x1, 128, u1h, 512, iw1i, iw1w, 4096, Au2h);
  launch_gemm(Au2h, 640, up2h, 640, up2b, nullptr, nullptr, 4096, 256, 640, u2h, 256, nullptr, 0, 0, 0, stream);

  // -------- FP: interp pos1->pos, seg head
  interp_idx_kernel<<<(16384 + 255) / 256, 256, 0, stream>>>(pos, 16384, pos1, 4096, iw0i, iw0w);
  build_interpA_kernel<<<(16384 * 256 + 255) / 256, 256, 0, stream>>>(nullptr, 0, u2h, 256, iw0i, iw0w, 16384, Afh);
  launch_gemm(Afh, 256, sg1h, 256, seg1b, alphaB, betaB, 16384, 512, 256, hh, 512, nullptr, 0, 1, 0, stream);
  // logits (13 x 16384) written transposed straight into d_out
  launch_gemm(hh, 512, sg2h, 512, seg2b, nullptr, nullptr, 16384, 13, 512, nullptr, 0,
              (float*)d_out, 16384, 0, 1, stream);
}
